// FilterDetections_65429531787961
// MI455X (gfx1250) — 8-bit matrix operand order repaired and checked by reading, compiled, not run on hardware
//
#include <hip/hip_runtime.h>
#include <hip/hip_bf16.h>

#define SCORE_T 0.05f
#define NMS_T   0.5f
#define MAXDET  300
#define KPAD    320
#define NBINS   256
#define TKCAP   4096
#define GCAP    32768

typedef int v8i __attribute__((ext_vector_type(8)));

struct SmemA {
    unsigned long long keys[TKCAP];
    unsigned int hist[NBINS];
    int thrBin;
    int cnt;
};
struct SmemB {
    float4 box[KPAD];
    int counts[16];
    unsigned char adj[KPAD * KPAD];
    unsigned char kept[KPAD];
    unsigned char valid[KPAD];
};
struct SmemC {
    unsigned long long keys[GCAP];
};

__device__ __forceinline__ unsigned long long pack_key(float s, unsigned int payload) {
    unsigned int sb = __float_as_uint(s) | 0x80000000u;
    return ((unsigned long long)sb << 32) | (unsigned int)(~payload);
}

__global__ void topk_per_class(const float* __restrict__ cls, int N, int C,
                               float* __restrict__ top_s, int* __restrict__ top_idx) {
    extern __shared__ char smem_raw[];
    SmemA& sm = *(SmemA*)smem_raw;
    const int tid = threadIdx.x;
    const int blk = blockIdx.x;
    const long long base = (long long)(blk / C) * N * C + (blk % C);

    for (int i = tid; i < NBINS; i += blockDim.x) sm.hist[i] = 0u;
    if (tid == 0) sm.cnt = 0;
    __syncthreads();

    const float binScale = (float)NBINS / (1.0f - SCORE_T);
    for (int n = tid; n < N; n += blockDim.x) {
        int np = n + 4 * (int)blockDim.x;
        if (np < N) __builtin_prefetch(&cls[base + (long long)np * C], 0, 1);
        float s = cls[base + (long long)n * C];
        if (s > SCORE_T) {
            int bin = (int)((s - SCORE_T) * binScale);
            bin = bin > (NBINS - 1) ? (NBINS - 1) : (bin < 0 ? 0 : bin);
            atomicAdd(&sm.hist[bin], 1u);
        }
    }
    __syncthreads();
    if (tid == 0) {
        unsigned int cum = 0; int thr = 0;
        for (int i = NBINS - 1; i >= 0; --i) {
            cum += sm.hist[i];
            if (cum >= MAXDET) { thr = i; break; }
        }
        sm.thrBin = thr;
    }
    __syncthreads();
    const int thr = sm.thrBin;

    for (int n = tid; n < N; n += blockDim.x) {
        int np = n + 4 * (int)blockDim.x;
        if (np < N) __builtin_prefetch(&cls[base + (long long)np * C], 0, 1);
        float s = cls[base + (long long)n * C];
        if (s > SCORE_T) {
            int bin = (int)((s - SCORE_T) * binScale);
            bin = bin > (NBINS - 1) ? (NBINS - 1) : (bin < 0 ? 0 : bin);
            if (bin >= thr) {
                int pos = atomicAdd(&sm.cnt, 1);
                if (pos < TKCAP) sm.keys[pos] = pack_key(s, (unsigned int)n);
            }
        }
    }
    __syncthreads();
    int nc = sm.cnt; if (nc > TKCAP) nc = TKCAP;
    for (int i = tid; i < TKCAP; i += blockDim.x)
        if (i >= nc) sm.keys[i] = 0ull;

    for (int k = 2; k <= TKCAP; k <<= 1)
        for (int j = k >> 1; j > 0; j >>= 1) {
            __syncthreads();
            for (int t = tid; t < TKCAP; t += blockDim.x) {
                int ixj = t ^ j;
                if (ixj > t) {
                    unsigned long long a = sm.keys[t], b = sm.keys[ixj];
                    bool doswap = ((t & k) == 0) ? (a < b) : (a > b);
                    if (doswap) { sm.keys[t] = b; sm.keys[ixj] = a; }
                }
            }
        }
    __syncthreads();

    for (int kk = tid; kk < KPAD; kk += blockDim.x) {
        float s; int idx;
        if (kk < MAXDET && kk < nc) {
            unsigned long long key = sm.keys[kk];
            s = __uint_as_float((unsigned int)(key >> 32) ^ 0x80000000u);
            idx = (int)(~(unsigned int)key);
        } else { s = -INFINITY; idx = 0; }
        top_s  [(long long)blk * KPAD + kk] = s;
        top_idx[(long long)blk * KPAD + kk] = idx;
    }
}

__global__ void nms_per_class(const float* __restrict__ boxes, int N, int C,
                              const float* __restrict__ top_s,
                              const int* __restrict__ top_idx,
                              unsigned char* __restrict__ keptg) {
    extern __shared__ char smem_raw[];
    SmemB& sm = *(SmemB*)smem_raw;
    const int tid = threadIdx.x;
    const int blk = blockIdx.x;
    const int b = blk / C;
    const float4* bx = (const float4*)boxes + (long long)b * N;

    for (int k = tid; k < KPAD; k += blockDim.x) {
        float s = top_s[(long long)blk * KPAD + k];
        int v = (s > 0.0f) ? 1 : 0;
        int idx = top_idx[(long long)blk * KPAD + k];
        sm.box[k]  = v ? bx[idx] : make_float4(0.f, 0.f, 0.f, 0.f);
        sm.valid[k] = (unsigned char)v;
        sm.kept[k]  = 0;
    }
    __syncthreads();

    for (int t = tid; t < KPAD * KPAD; t += blockDim.x) {
        int i = t / KPAD, j = t - i * KPAD;
        float4 A = sm.box[i], B2 = sm.box[j];
        float ltx = fmaxf(A.x, B2.x), lty = fmaxf(A.y, B2.y);
        float rbx = fminf(A.z, B2.z), rby = fminf(A.w, B2.w);
        float inter = fmaxf(rbx - ltx, 0.f) * fmaxf(rby - lty, 0.f);
        float a1 = (A.z - A.x) * (A.w - A.y);
        float a2 = (B2.z - B2.x) * (B2.w - B2.y);
        float un = fmaxf(a1 + a2 - inter, 1e-7f);
        sm.adj[t] = (inter / un >= NMS_T) ? (unsigned char)1 : (unsigned char)0;
    }
    __syncthreads();

    if (tid < 32) {
        const int lane = tid;
        const int rsel = lane & 15;
        const int hiA = (lane >> 4) ? 8 : 0;
        const int hiB = (lane >> 4) ? 16 : 0;
        for (int t = 0; t < KPAD / 16; ++t) {
            const int m0 = t * 16;
            const int* arow = (const int*)(sm.adj + (long long)(m0 + rsel) * KPAD);
            const int* kw = (const int*)sm.kept;
            v8i acc = {};
            for (int g = 0; g < KPAD / 64; ++g) {
                const int j0 = g * 64;
                const int oa = (j0 + hiA) >> 2;
                v8i a;
                a[0] = arow[oa + 0];  a[1] = arow[oa + 1];
                a[2] = arow[oa + 4];  a[3] = arow[oa + 5];
                a[4] = arow[oa + 8];  a[5] = arow[oa + 9];
                a[6] = arow[oa + 12]; a[7] = arow[oa + 13];
                const int ob = (j0 + hiA) >> 2;
                const int ob2 = (j0 + 32 + hiB) >> 2;
                v8i bm;
                bm[0] = kw[ob + 0];  bm[1] = kw[ob + 1];
                bm[2] = kw[ob + 4];  bm[3] = kw[ob + 5];
                bm[4] = kw[ob + 8];  bm[5] = kw[ob + 9];
                bm[6] = kw[ob + 12]; bm[7] = kw[ob + 13];
                acc = __builtin_amdgcn_wmma_i32_16x16x64_iu8(false, a, false, bm,
                                                             acc, false, false);
            }
            if (lane == 0)       { for (int r = 0; r < 8; ++r) sm.counts[r]     = acc[r]; }
            else if (lane == 16) { for (int r = 0; r < 8; ++r) sm.counts[8 + r] = acc[r]; }
            asm volatile("s_wait_dscnt 0" ::: "memory");
            if (lane == 0) {
                for (int i2 = 0; i2 < 16; ++i2) {
                    const int i = m0 + i2;
                    int supp = (sm.counts[i2] > 0) ? 1 : 0;
                    for (int p = 0; p < i2; ++p)
                        supp |= (int)(sm.kept[m0 + p] & sm.adj[(long long)i * KPAD + m0 + p]);
                    sm.kept[i] = (unsigned char)(sm.valid[i] & (supp ? 0 : 1));
                }
            }
            asm volatile("s_wait_dscnt 0" ::: "memory");
            __builtin_amdgcn_wave_barrier();
        }
    }
    __syncthreads();
    for (int k = tid; k < KPAD; k += blockDim.x)
        keptg[(long long)blk * KPAD + k] = sm.kept[k];
}

__global__ void global_topk(const float* __restrict__ boxes, int N, int C,
                            const float* __restrict__ top_s,
                            const int* __restrict__ top_idx,
                            const unsigned char* __restrict__ keptg,
                            float* __restrict__ out_boxes,
                            float* __restrict__ out_scores,
                            int* __restrict__ out_labels) {
    extern __shared__ char smem_raw[];
    SmemC& sm = *(SmemC*)smem_raw;
    const int tid = threadIdx.x;
    const int b = blockIdx.x;
    const int M = C * MAXDET;

    for (int t = tid; t < GCAP; t += blockDim.x) {
        unsigned long long key = 0ull;
        if (t < M) {
            int c = t / MAXDET, k = t - c * MAXDET;
            long long off = (long long)(b * C + c) * KPAD + k;
            float s = top_s[off];
            if (keptg[off] && s > 0.0f) key = pack_key(s, (unsigned int)t);
        }
        sm.keys[t] = key;
    }
    for (int k = 2; k <= GCAP; k <<= 1)
        for (int j = k >> 1; j > 0; j >>= 1) {
            __syncthreads();
            for (int t = tid; t < GCAP; t += blockDim.x) {
                int ixj = t ^ j;
                if (ixj > t) {
                    unsigned long long a = sm.keys[t], bb = sm.keys[ixj];
                    bool doswap = ((t & k) == 0) ? (a < bb) : (a > bb);
                    if (doswap) { sm.keys[t] = bb; sm.keys[ixj] = a; }
                }
            }
        }
    __syncthreads();

    for (int k = tid; k < MAXDET; k += blockDim.x) {
        unsigned long long key = sm.keys[k];
        float4* ob4 = (float4*)(out_boxes + ((long long)b * MAXDET + k) * 4);
        if (key >> 32) {
            int t = (int)(~(unsigned int)key);
            int c = t / MAXDET, kk = t - c * MAXDET;
            float s = __uint_as_float((unsigned int)(key >> 32) ^ 0x80000000u);
            int idx = top_idx[(long long)(b * C + c) * KPAD + kk];
            *ob4 = ((const float4*)boxes)[(long long)b * N + idx];
            out_scores[b * MAXDET + k] = s;
            out_labels[b * MAXDET + k] = c;
        } else {
            *ob4 = make_float4(-1.f, -1.f, -1.f, -1.f);
            out_scores[b * MAXDET + k] = -1.0f;
            out_labels[b * MAXDET + k] = -1;
        }
    }
}

extern "C" void kernel_launch(void* const* d_in, const int* in_sizes, int n_in,
                              void* d_out, int out_size, void* d_ws, size_t ws_size,
                              hipStream_t stream) {
    (void)n_in; (void)ws_size;
    const float* boxes = (const float*)d_in[0];
    const float* cls   = (const float*)d_in[1];

    const int B = out_size / (MAXDET * 6);
    const long long BN = (long long)in_sizes[0] / 4;
    const int N = (int)(BN / B);
    const int C = (int)((long long)in_sizes[1] / BN);

    float* top_s = (float*)d_ws;
    int* top_idx = (int*)((char*)d_ws + (size_t)B * C * KPAD * 4);
    unsigned char* keptg = (unsigned char*)((char*)d_ws + (size_t)B * C * KPAD * 8);

    float* out = (float*)d_out;
    float* out_boxes  = out;
    float* out_scores = out + (size_t)B * MAXDET * 4;
    int*   out_labels = (int*)(out + (size_t)B * MAXDET * 5);

    topk_per_class<<<dim3(B * C), dim3(256), sizeof(SmemA), stream>>>(
        cls, N, C, top_s, top_idx);
    nms_per_class<<<dim3(B * C), dim3(256), sizeof(SmemB), stream>>>(
        boxes, N, C, top_s, top_idx, keptg);
    global_topk<<<dim3(B), dim3(1024), sizeof(SmemC), stream>>>(
        boxes, N, C, top_s, top_idx, keptg, out_boxes, out_scores, out_labels);
}
